// CausalSelfAttention_26164940767964
// MI455X (gfx1250) — matrix operand order repaired and checked by reading, not compiled, not run on hardware
//
#include <hip/hip_runtime.h>
#include <stdint.h>

typedef _Float16 h16_t;
typedef __attribute__((ext_vector_type(16))) _Float16 h16x16;
typedef __attribute__((ext_vector_type(8)))  _Float16 h16x8;
typedef __attribute__((ext_vector_type(8)))  float    f32x8;

__device__ __forceinline__ f32x8 wmma_f16(h16x16 a, h16x16 b, f32x8 c) {
  return __builtin_amdgcn_wmma_f32_16x16x32_f16(false, a, false, b, (short)0, c,
                                                false, false);
}

// Async copy 16B global -> LDS (ASYNCcnt-tracked). Generic pointers into
// __shared__ carry the group-segment offset in addr[31:0] (LDS aperture),
// which is exactly the VDST LDS-offset operand of the async op.
__device__ __forceinline__ void async_load_b128(const void* gsrc, void* ldst) {
  uint32_t lds = (uint32_t)(uintptr_t)ldst;
  uint64_t ga = (uint64_t)(uintptr_t)gsrc;
  asm volatile("global_load_async_to_lds_b128 %0, %1, off"
               :
               : "v"(lds), "v"(ga)
               : "memory");
}
__device__ __forceinline__ void wait_asynccnt0() {
  asm volatile("s_wait_asynccnt 0" ::: "memory");
}

// ---------------------------------------------------------------------------
// Prep: softmax(alpha_embed), softmax(alpha_heads), suffix sums, combo coefs.
// sbuf[0..2] = suffix sums of ae for buckets (<256, <512, <1024)
// sbuf[3+hi*3+ei] = ah[hi]*ae[ei]
// ---------------------------------------------------------------------------
__global__ void prep_kernel(const float* __restrict__ aE,
                            const float* __restrict__ aH,
                            float* __restrict__ sbuf) {
  if (threadIdx.x == 0 && blockIdx.x == 0) {
    float e0 = aE[0], e1 = aE[1], e2 = aE[2];
    float me = fmaxf(e0, fmaxf(e1, e2));
    float x0 = __expf(e0 - me), x1 = __expf(e1 - me), x2 = __expf(e2 - me);
    float se = x0 + x1 + x2;
    float ae[3] = {x0 / se, x1 / se, x2 / se};
    float h0 = aH[0], h1 = aH[1], h2 = aH[2];
    float mh = fmaxf(h0, fmaxf(h1, h2));
    float y0 = __expf(h0 - mh), y1 = __expf(h1 - mh), y2 = __expf(h2 - mh);
    float sh = y0 + y1 + y2;
    float ah[3] = {y0 / sh, y1 / sh, y2 / sh};
    sbuf[0] = ae[0] + ae[1] + ae[2];
    sbuf[1] = ae[1] + ae[2];
    sbuf[2] = ae[2];
    for (int hi = 0; hi < 3; hi++)
      for (int ei = 0; ei < 3; ei++) sbuf[3 + hi * 3 + ei] = ah[hi] * ae[ei];
  }
}

// ---------------------------------------------------------------------------
// fp32 -> f16 conversion (vectorized x4)
// ---------------------------------------------------------------------------
__global__ __launch_bounds__(256) void f32_to_f16_kernel(
    const float* __restrict__ in, h16_t* __restrict__ out, int n) {
  int idx = (blockIdx.x * blockDim.x + threadIdx.x) * 4;
  if (idx + 3 < n) {
    float4 v = *(const float4*)(in + idx);
    out[idx + 0] = (h16_t)v.x;
    out[idx + 1] = (h16_t)v.y;
    out[idx + 2] = (h16_t)v.z;
    out[idx + 3] = (h16_t)v.w;
  }
}

// ---------------------------------------------------------------------------
// GEMM:  Out[m,o] = sum_c A[m,c] * s(max(o%rmod, c)) * W[o,c]
// A: f16 row-major M x K (async-staged to LDS).  W: fp32 row-major N x K
// (scaled+converted in VALU during staging).  Block tile 128x128, K-step 32,
// 8 waves, each wave 2x4 WMMA frags.
// ---------------------------------------------------------------------------
#define GSTR 40  // LDS row stride in f16 elems (80B: 16B-aligned, conflict-light)

template <bool OUT_H16>
__global__ __launch_bounds__(256) void gemm_mix_kernel(
    const h16_t* __restrict__ A, const float* __restrict__ W,
    const float* __restrict__ sbuf, void* __restrict__ outp, int M, int N,
    int K, int rmod) {
  __shared__ __align__(16) h16_t Al[128 * GSTR];
  __shared__ __align__(16) h16_t Bl[128 * GSTR];
  const int tid = threadIdx.x;
  const int lane = tid & 31;
  const int wid = tid >> 5;
  const int wm = wid & 3;   // 4 row-groups of 32
  const int wn = wid >> 2;  // 2 col-groups of 64
  const int m0 = blockIdx.x * 128;
  const int n0 = blockIdx.y * 128;
  const float s0v = sbuf[0], s1v = sbuf[1], s2v = sbuf[2];

  f32x8 acc[2][4];
#pragma unroll
  for (int i = 0; i < 2; i++)
#pragma unroll
    for (int j = 0; j < 4; j++)
#pragma unroll
      for (int r = 0; r < 8; r++) acc[i][j][r] = 0.f;

  const int arow = tid >> 1;        // 0..127
  const int acol = (tid & 1) * 16;  // 0 or 16
  const int l15 = lane & 15;
  const int rglob = n0 + arow;
  const int rm_w = rglob % rmod;

  for (int k0 = 0; k0 < K; k0 += 32) {
    {  // stage A tile (128 x 32 f16) asynchronously into LDS
      const h16_t* src = A + (size_t)(m0 + arow) * K + k0 + acol;
      async_load_b128(src, &Al[arow * GSTR + acol]);
      async_load_b128(src + 8, &Al[arow * GSTR + acol + 8]);
    }
    if (k0 + 32 < K) {  // prefetch next K-step sources
      __builtin_prefetch(A + (size_t)(m0 + arow) * K + k0 + 32 + acol, 0, 1);
      __builtin_prefetch(W + (size_t)rglob * K + k0 + 32 + acol, 0, 1);
    }
    {  // stage W tile (128 o x 32 c), scale by mixed-alpha suffix sums
      const float* src = W + (size_t)rglob * K + k0 + acol;
      union { h16x8 v[2]; h16_t e[16]; } tb;
#pragma unroll
      for (int j = 0; j < 16; j++) {
        int c = k0 + acol + j;
        int mx = rm_w > c ? rm_w : c;
        float s = mx < 256 ? s0v : (mx < 512 ? s1v : s2v);
        tb.e[j] = (h16_t)(src[j] * s);
      }
      *(h16x8*)&Bl[arow * GSTR + acol] = tb.v[0];
      *(h16x8*)&Bl[arow * GSTR + acol + 8] = tb.v[1];
    }
    wait_asynccnt0();
    __syncthreads();

    h16x16 af[2], bfm[4];
    {  // A frags: lane -> M=lane%16; elems i -> K = i + 16*(lane>=16)
      int kb = (lane & 16) ? 16 : 0;
#pragma unroll
      for (int i = 0; i < 2; i++) {
        int row = wm * 32 + i * 16 + l15;
        const uint32_t* p0 = (const uint32_t*)&Al[row * GSTR + kb];
        const uint32_t* p1 = (const uint32_t*)&Al[row * GSTR + 8 + kb];
        union { h16x16 v; uint32_t u[8]; } u;
#pragma unroll
        for (int j = 0; j < 4; j++) { u.u[j] = p0[j]; u.u[4 + j] = p1[j]; }
        af[i] = u.v;
      }
    }
    {  // B frags: lane -> N=lane%16; elems i -> K = i + 16*(lane>=16)
      int kb = (lane & 16) ? 16 : 0;
#pragma unroll
      for (int j = 0; j < 4; j++) {
        int col = wn * 64 + j * 16 + l15;
        const uint32_t* p = (const uint32_t*)&Bl[col * GSTR + kb];
        union { h16x16 v; uint32_t u[8]; } u;
#pragma unroll
        for (int t = 0; t < 8; t++) u.u[t] = p[t];
        bfm[j] = u.v;
      }
    }
#pragma unroll
    for (int i = 0; i < 2; i++)
#pragma unroll
      for (int j = 0; j < 4; j++) acc[i][j] = wmma_f16(af[i], bfm[j], acc[i][j]);
    __syncthreads();
  }

  // epilogue: C layout VGPR r -> M = r + 8*(lane>=16), N = lane%16
  const int h16off = (lane & 16) ? 8 : 0;
#pragma unroll
  for (int i = 0; i < 2; i++)
#pragma unroll
    for (int j = 0; j < 4; j++)
#pragma unroll
      for (int r = 0; r < 8; r++) {
        size_t row = m0 + wm * 32 + i * 16 + r + h16off;
        size_t col = n0 + wn * 64 + j * 16 + l15;
        float v = acc[i][j][r];
        if (OUT_H16)
          ((h16_t*)outp)[row * N + col] = (h16_t)v;
        else
          ((float*)outp)[row * N + col] = v;
      }
}

// ---------------------------------------------------------------------------
// Flash attention for one (embed e, heads h) combo, head dim D = e/h.
// Block = 4 waves x 32, one (b, head, 64-row q tile). Each wave owns 16 q rows.
// K and V tiles async-copied to LDS (coalesced b128); V transposed LDS->LDS.
// Accumulates coef * softmax(QK^T/sqrt(D)) V  into fp32 y (race-free layout).
// ---------------------------------------------------------------------------
template <int D>
__global__ __launch_bounds__(128) void flash_kernel(
    const h16_t* __restrict__ qkv, const float* __restrict__ sbuf, int coefIdx,
    float* __restrict__ y) {
  constexpr int KP = (D < 32) ? 32 : D;  // contraction dim padded to WMMA K
  constexpr int NDC = KP / 32;           // Q/K 32-chunks
  constexpr int NDN = D / 16;            // O column frags
  constexpr int KSTR = KP + 8;
  constexpr int VSTR = 40;  // 32 keys + pad
  constexpr int LG = (D == 16) ? 4 : (D == 32) ? 5 : (D == 64) ? 6
                                                : (D == 128) ? 7 : 8;
  __shared__ __align__(16) h16_t Klds[32 * KSTR];  // [key][c]
  __shared__ __align__(16) h16_t Vraw[32 * D];     // [key][c] row-major
  __shared__ __align__(16) h16_t Vt[D * VSTR];     // [c][key] (transposed)
  __shared__ __align__(16) h16_t Plds[4 * 16 * 32];

  const int tid = threadIdx.x, lane = tid & 31, wv = tid >> 5;
  const int l15 = lane & 15;
  const int h16off = (lane & 16) ? 8 : 0;
  const int tile = blockIdx.x * 64;
  const int head = blockIdx.y;
  const int b = blockIdx.z;
  const int T = 1024, C3 = 3072;
  const float coef = sbuf[3 + coefIdx];
  const float isd = rsqrtf((float)D);
  const int qbase = tile + wv * 16;

  // D==16: zero Klds once so the padded K columns (16..31) stay zero forever.
  if (D < 32) {
    for (int idx = tid; idx < 32 * KSTR; idx += 128) Klds[idx] = (h16_t)0.f;
  }

  // Q A-frags (row = qbase + lane%16)
  h16x16 qf[NDC];
  {
    const h16_t* qp = qkv + (size_t)(b * T + qbase + l15) * C3 + head * D;
    int kb = (lane & 16) ? 16 : 0;
#pragma unroll
    for (int dc = 0; dc < NDC; dc++) {
      h16x16 v;
#pragma unroll
      for (int i = 0; i < 16; i++) {
        int c = dc * 32 + i + kb;
        v[i] = (c < D) ? qp[c] : (h16_t)0.f;
      }
      qf[dc] = v;
    }
  }

  float m_i[8], l_i[8];
#pragma unroll
  for (int r = 0; r < 8; r++) { m_i[r] = -1e30f; l_i[r] = 0.f; }
  f32x8 oacc[NDN];
#pragma unroll
  for (int j = 0; j < NDN; j++)
#pragma unroll
    for (int r = 0; r < 8; r++) oacc[j][r] = 0.f;

  for (int kt = 0; kt < tile + 64; kt += 32) {
    __syncthreads();  // prior-iteration consumers done (also covers D<32 init)
    // async stage: K rows and V rows (coalesced 16B chunks, no VGPR bounce)
    for (int idx = tid; idx < 32 * (D / 8); idx += 128) {
      int key = idx / (D / 8);
      int c8 = (idx % (D / 8)) * 8;
      const h16_t* base = qkv + (size_t)(b * T + kt + key) * C3 + head * D + c8;
      async_load_b128(base + 1024, &Klds[key * KSTR + c8]);  // K
      async_load_b128(base + 2048, &Vraw[key * D + c8]);     // V (row-major)
    }
    wait_asynccnt0();
    __syncthreads();
    // transpose V in LDS: Vt[c][key] = Vraw[key][c] (contiguous reads per lane)
    for (int idx = tid; idx < 32 * D; idx += 128) {
      int c = idx & (D - 1);
      int key = idx >> LG;
      Vt[c * VSTR + key] = Vraw[key * D + c];
    }
    __syncthreads();

    // S = Q K^T for two 16-key column tiles
    f32x8 sfr[2];
#pragma unroll
    for (int n16 = 0; n16 < 2; n16++) {
      f32x8 sacc;
#pragma unroll
      for (int r = 0; r < 8; r++) sacc[r] = 0.f;
      int key = n16 * 16 + l15;
      int kbb = (lane & 16) ? 16 : 0;
#pragma unroll
      for (int dc = 0; dc < NDC; dc++) {
        const uint32_t* p = (const uint32_t*)&Klds[key * KSTR + dc * 32 + kbb];
        union { h16x16 v; uint32_t u[8]; } u;
#pragma unroll
        for (int t = 0; t < 8; t++) u.u[t] = p[t];
        sacc = wmma_f16(qf[dc], u.v, sacc);
      }
      sfr[n16] = sacc;
    }

    // causal mask + scale, online softmax (16-lane butterfly per half)
    float rmax[8];
#pragma unroll
    for (int r = 0; r < 8; r++) {
      int q = qbase + r + h16off;
      float v0 = (kt + l15 <= q) ? sfr[0][r] * isd : -1e30f;
      float v1 = (kt + 16 + l15 <= q) ? sfr[1][r] * isd : -1e30f;
      sfr[0][r] = v0;
      sfr[1][r] = v1;
      rmax[r] = fmaxf(v0, v1);
    }
#pragma unroll
    for (int off = 1; off < 16; off <<= 1)
#pragma unroll
      for (int r = 0; r < 8; r++)
        rmax[r] = fmaxf(rmax[r], __shfl_xor(rmax[r], off, 32));
    float alpha[8], rsum[8];
#pragma unroll
    for (int r = 0; r < 8; r++) {
      float mn = fmaxf(m_i[r], rmax[r]);
      alpha[r] = __expf(m_i[r] - mn);
      m_i[r] = mn;
      float e0 = __expf(sfr[0][r] - mn);
      float e1 = __expf(sfr[1][r] - mn);
      sfr[0][r] = e0;
      sfr[1][r] = e1;
      rsum[r] = e0 + e1;
    }
#pragma unroll
    for (int off = 1; off < 16; off <<= 1)
#pragma unroll
      for (int r = 0; r < 8; r++) rsum[r] += __shfl_xor(rsum[r], off, 32);
#pragma unroll
    for (int r = 0; r < 8; r++) {
      l_i[r] = l_i[r] * alpha[r] + rsum[r];
#pragma unroll
      for (int j = 0; j < NDN; j++) oacc[j][r] *= alpha[r];
    }

    // repack P (C-frag layout) -> A-frag layout via per-wave LDS scratch
    {
      h16_t* P = &Plds[wv * 16 * 32];
#pragma unroll
      for (int r = 0; r < 8; r++) {
        int row = r + h16off;
        P[row * 32 + l15] = (h16_t)sfr[0][r];
        P[row * 32 + 16 + l15] = (h16_t)sfr[1][r];
      }
      asm volatile("s_wait_dscnt 0" ::: "memory");  // wave-private LDS RAW
      h16x16 pf;
      {
        int kb = (lane & 16) ? 16 : 0;
        const uint32_t* p0 = (const uint32_t*)&P[l15 * 32 + kb];
        const uint32_t* p1 = (const uint32_t*)&P[l15 * 32 + 8 + kb];
        union { h16x16 v; uint32_t u[8]; } u;
#pragma unroll
        for (int j = 0; j < 4; j++) { u.u[j] = p0[j]; u.u[4 + j] = p1[j]; }
        pf = u.v;
      }
      // O += P V   (B frag from transposed V: contiguous 16 keys per lane)
      int kbb = (lane & 16) ? 16 : 0;
#pragma unroll
      for (int j = 0; j < NDN; j++) {
        const uint32_t* p = (const uint32_t*)&Vt[(j * 16 + l15) * VSTR + kbb];
        union { h16x16 v; uint32_t u[8]; } u;
#pragma unroll
        for (int t = 0; t < 8; t++) u.u[t] = p[t];
        oacc[j] = wmma_f16(pf, u.v, oacc[j]);
      }
    }
  }

  // y[b, q, head*D + col] += coef * O / l   (disjoint writers in this launch)
#pragma unroll
  for (int r = 0; r < 8; r++) {
    int q = qbase + r + h16off;
    float inv = coef / l_i[r];
    float* yrow = y + (size_t)(b * T + q) * 1024 + head * D;
#pragma unroll
    for (int j = 0; j < NDN; j++) {
      int col = j * 16 + l15;
      yrow[col] += oacc[j][r] * inv;
    }
  }
}

static void launch_flash(int D, int h, int coefIdx, const h16_t* qkv,
                         const float* sbuf, float* y, hipStream_t stream) {
  dim3 g(1024 / 64, h, 4);
  switch (D) {
    case 16:  flash_kernel<16><<<g, 128, 0, stream>>>(qkv, sbuf, coefIdx, y); break;
    case 32:  flash_kernel<32><<<g, 128, 0, stream>>>(qkv, sbuf, coefIdx, y); break;
    case 64:  flash_kernel<64><<<g, 128, 0, stream>>>(qkv, sbuf, coefIdx, y); break;
    case 128: flash_kernel<128><<<g, 128, 0, stream>>>(qkv, sbuf, coefIdx, y); break;
    case 256: flash_kernel<256><<<g, 128, 0, stream>>>(qkv, sbuf, coefIdx, y); break;
  }
}

// ---------------------------------------------------------------------------
extern "C" void kernel_launch(void* const* d_in, const int* in_sizes, int n_in,
                              void* d_out, int out_size, void* d_ws,
                              size_t ws_size, hipStream_t stream) {
  (void)in_sizes; (void)n_in; (void)out_size; (void)ws_size;
  const float* x = (const float*)d_in[0];
  // d_in[1] = i (unused by reference)
  const float* aE = (const float*)d_in[2];
  const float* aH = (const float*)d_in[3];
  const float* W_attn = (const float*)d_in[4];
  const float* W_proj = (const float*)d_in[5];
  float* out = (float*)d_out;

  const size_t MB = 1024 * 1024;
  char* ws = (char*)d_ws;
  float* sbuf = (float*)ws;                    // 256 B
  h16_t* xhf = (h16_t*)(ws + 256);             // 8 MB
  h16_t* qkvh = (h16_t*)(ws + 256 + 8 * MB);   // 24 MB
  float* yacc = (float*)(ws + 256 + 32 * MB);  // 16 MB
  h16_t* yhf = (h16_t*)(ws + 256 + 48 * MB);   // 8 MB

  prep_kernel<<<1, 32, 0, stream>>>(aE, aH, sbuf);

  {  // x -> f16
    int n = 4 * 1024 * 1024;
    f32_to_f16_kernel<<<n / 4 / 256, 256, 0, stream>>>(x, xhf, n);
  }

  // qkv = x @ Wmix_attn^T  (4096 x 3072), scale fused into B staging
  gemm_mix_kernel<true><<<dim3(4096 / 128, 3072 / 128), 256, 0, stream>>>(
      xhf, W_attn, sbuf, (void*)qkvh, 4096, 3072, 1024, 1024);

  hipMemsetAsync(yacc, 0, (size_t)4096 * 1024 * 4, stream);

  // 9 attention combos, accumulating into yacc
  const int HS[3] = {4, 8, 16};
  const int ES[3] = {256, 512, 1024};
  for (int hi = 0; hi < 3; hi++)
    for (int ei = 0; ei < 3; ei++)
      launch_flash(ES[ei] / HS[hi], HS[hi], hi * 3 + ei, qkvh, sbuf, yacc,
                   stream);

  {  // y -> f16
    int n = 4096 * 1024;
    f32_to_f16_kernel<<<n / 4 / 256, 256, 0, stream>>>(yacc, yhf, n);
  }

  // out = y @ Wmix_proj^T  (4096 x 1024), fp32 output
  gemm_mix_kernel<false><<<dim3(4096 / 128, 1024 / 128), 256, 0, stream>>>(
      yhf, W_proj, sbuf, (void*)out, 4096, 1024, 1024, 1024);
}
